// DyIntraModalityUpdate_78812649882399
// MI455X (gfx1250) — compile-verified, matrix operand order checked by reading, not run on hardware
//
#include <hip/hip_runtime.h>
#include <stdint.h>

#define B_    32
#define NSEQ  512
#define D_    1024
#define H_    16
#define N3    3072
#define MROWS (B_ * NSEQ)   // 16384

typedef __bf16 bf16x16 __attribute__((ext_vector_type(16)));
typedef __bf16 bf16x2  __attribute__((ext_vector_type(2)));
typedef float  f32x8   __attribute__((ext_vector_type(8)));

__device__ __forceinline__ unsigned short f2bf(float f) {
  __bf16 h = (__bf16)f;                       // native cvt
  return __builtin_bit_cast(unsigned short, h);
}
__device__ __forceinline__ float bf2f(unsigned short h) {
  union { unsigned u; float f; } x; x.u = ((unsigned)h) << 16;
  return x.f;
}
__device__ __forceinline__ unsigned pack2(float a, float b) {
  bf16x2 p; p[0] = (__bf16)a; p[1] = (__bf16)b;   // v_cvt_pk_bf16_f32
  return __builtin_bit_cast(unsigned, p);
}

// Load one WMMA A/B bf16 fragment from LDS. Per CDNA5 ISA layout:
// lanes 0-15 hold K{0..7} and K{16..23}; lanes 16-31 hold K{8..15} and K{24..31}.
__device__ __forceinline__ bf16x16 ld_frag(const char* p) {
  union { uint4 u[2]; bf16x16 v; } r;
  r.u[0] = *(const uint4*)(p);
  r.u[1] = *(const uint4*)(p + 32);
  return r.v;
}

__device__ __forceinline__ void zero8(f32x8& a) {
  #pragma unroll
  for (int e = 0; e < 8; e++) a[e] = 0.f;
}

// ---- CDNA5 async global->LDS copy (ASYNCcnt path, no VGPR staging) ----------
__device__ __forceinline__ void async_ld_b128(void* lds, const void* g) {
  unsigned off = (unsigned)(uintptr_t)lds;     // low 32 bits = LDS byte address
  asm volatile("global_load_async_to_lds_b128 %0, %1, off"
               :: "v"(off), "v"(g) : "memory");
}
__device__ __forceinline__ void wait_async() {
  asm volatile("s_wait_asynccnt 0x0" ::: "memory");
}

// Issue one 128x32 A tile + 128-col B tile fill (16B chunks per lane).
__device__ __forceinline__ void gemm_issue_tile(
    const unsigned* __restrict__ Xd, const unsigned* __restrict__ Wt,
    unsigned* sAbuf, unsigned* sBbuf, int t, int m0, int n0, int kk)
{
  #pragma unroll
  for (int s = 0; s < 2; s++) {                // 512 chunks of 16B each
    int q = t + s * 256;
    int row = q >> 2, c4 = (q & 3) * 4;
    async_ld_b128(&sAbuf[row * 16 + c4],
                  Xd + (size_t)(m0 + row) * 512 + (kk >> 1) + c4);
  }
  #pragma unroll
  for (int s = 0; s < 2; s++) {
    int q = t + s * 256;
    int col = q >> 2, c4 = (q & 3) * 4;
    async_ld_b128(&sBbuf[col * 16 + c4],
                  Wt + (size_t)(n0 + col) * 512 + (kk >> 1) + c4);
  }
}

// ------------------------------------------------ pre-pass: relu(x) -> bf16
__global__ __launch_bounds__(256) void cvt_relu_bf16(
    const float* __restrict__ X, unsigned* __restrict__ out /* dwords */)
{
  size_t base = ((size_t)blockIdx.x * 256 + threadIdx.x) * 8;
  uint4 d;
  float4 a = *(const float4*)(X + base);
  float4 b = *(const float4*)(X + base + 4);
  d.x = pack2(fmaxf(a.x, 0.f), fmaxf(a.y, 0.f));
  d.y = pack2(fmaxf(a.z, 0.f), fmaxf(a.w, 0.f));
  d.z = pack2(fmaxf(b.x, 0.f), fmaxf(b.y, 0.f));
  d.w = pack2(fmaxf(b.z, 0.f), fmaxf(b.w, 0.f));
  *(uint4*)(out + base / 2) = d;
}

// -------------------------- pre-pass: W f32 [1024][N] -> bf16 [n][kpair] dwords
__global__ __launch_bounds__(256) void transpose_w_bf16(
    const float* __restrict__ W, unsigned* __restrict__ Wt, int N)
{
  int n  = blockIdx.x * 256 + threadIdx.x;
  int kp = blockIdx.y;                       // 0..511
  float a = W[(size_t)(2 * kp)     * N + n];
  float b = W[(size_t)(2 * kp + 1) * N + n];
  Wt[(size_t)n * 512 + kp] = pack2(a, b);
}

// ---------------------------------------------------------------- mean pooling
__global__ __launch_bounds__(256) void mean_pool(
    const float* __restrict__ v, const float* __restrict__ t,
    const float* __restrict__ v_mask, const float* __restrict__ t_mask,
    float* __restrict__ mean_v, float* __restrict__ mean_t)
{
  int bid = blockIdx.x;          // 256 = 2 modalities * 32 b * 4 dblk
  int mod = bid >> 7;
  int rem = bid & 127;
  int b = rem >> 2;
  int d = (rem & 3) * 256 + threadIdx.x;
  const float* x = mod ? t : v;
  const float* m = mod ? t_mask : v_mask;
  float* o = mod ? mean_t : mean_v;
  float s = 0.f, ms = 0.f;
  for (int n = 0; n < NSEQ; n++) {
    float mm = m[b * NSEQ + n];
    s  += x[((size_t)b * NSEQ + n) * D_ + d] * mm;
    ms += mm;
  }
  o[b * D_ + d] = s / ms;
}

// ---------------------------------------------------------------- gate GEMM
__global__ __launch_bounds__(256) void gate_gemm(
    const float* __restrict__ mean_v, const float* __restrict__ mean_t,
    const float* __restrict__ W_v4t, const float* __restrict__ b_v4t,
    const float* __restrict__ W_t4v, const float* __restrict__ b_t4v,
    float* __restrict__ gate_v4t, float* __restrict__ gate_t4v)
{
  int bid = blockIdx.x;          // 256 blocks
  int mod = bid >> 7;            // 0: v4t(from mean_v), 1: t4v(from mean_t)
  int rem = bid & 127;
  int b = rem >> 2;
  int j = (rem & 3) * 256 + threadIdx.x;
  const float* mean = mod ? mean_t : mean_v;
  const float* W    = mod ? W_t4v  : W_v4t;
  const float* bb   = mod ? b_t4v  : b_v4t;
  float* o          = mod ? gate_t4v : gate_v4t;
  float acc = bb[j];
  for (int k = 0; k < D_; k++) {
    float a = mean[b * D_ + k];
    a = a > 0.f ? a : 0.f;
    acc += a * W[(size_t)k * D_ + j];
  }
  o[b * D_ + j] = 1.f / (1.f + __expf(-acc));
}

// ------------------------------------------- KQV GEMM (pre-relu'd bf16 inputs)
// C[m, j] = (Xb[m,:] @ W[:,j] + bias[j]) * mask[m]  -> bf16
// Xb: [M][1024] bf16 (relu applied);  Wt: [N3][512] kpair dwords (bf16)
__global__ __launch_bounds__(256) void kqv_gemm(
    const unsigned* __restrict__ Xd, const unsigned* __restrict__ Wt,
    const float* __restrict__ bias, const float* __restrict__ mask,
    unsigned short* __restrict__ out)
{
  __shared__ unsigned sA[2][128 * 16];   // [row][kpair] dwords, double buffered
  __shared__ unsigned sB[2][128 * 16];   // [col][kpair] dwords
  const int t = threadIdx.x;
  const int lane = t & 31, wv = t >> 5;
  const int half = lane >> 4, r = lane & 15;
  const int m0 = blockIdx.x * 128;
  const int n0 = blockIdx.y * 128;
  const int wm = (wv >> 2) * 64;   // 0/64
  const int wn = (wv & 3) * 32;    // 0..96
  f32x8 acc[4][2];
  #pragma unroll
  for (int i = 0; i < 4; i++)
    #pragma unroll
    for (int j = 0; j < 2; j++) zero8(acc[i][j]);

  gemm_issue_tile(Xd, Wt, sA[0], sB[0], t, m0, n0, 0);
  wait_async();
  __syncthreads();
  int cur = 0;
  for (int kk = 0; kk < D_; kk += 32) {
    if (kk + 32 < D_)                      // prefetch next chunk via async DMA
      gemm_issue_tile(Xd, Wt, sA[cur ^ 1], sB[cur ^ 1], t, m0, n0, kk + 32);
    bf16x16 aF[4], bF[2];
    #pragma unroll
    for (int mt = 0; mt < 4; mt++)
      aF[mt] = ld_frag((const char*)sA[cur] + (wm + mt * 16 + r) * 64 + half * 16);
    #pragma unroll
    for (int nt = 0; nt < 2; nt++)
      bF[nt] = ld_frag((const char*)sB[cur] + (wn + nt * 16 + r) * 64 + half * 16);
    #pragma unroll
    for (int mt = 0; mt < 4; mt++)
      #pragma unroll
      for (int nt = 0; nt < 2; nt++)
        acc[mt][nt] = __builtin_amdgcn_wmma_f32_16x16x32_bf16(
            false, aF[mt], false, bF[nt], (short)0, acc[mt][nt], false, false);
    wait_async();
    __syncthreads();
    cur ^= 1;
  }
  #pragma unroll
  for (int mt = 0; mt < 4; mt++)
    #pragma unroll
    for (int nt = 0; nt < 2; nt++)
      #pragma unroll
      for (int j = 0; j < 8; j++) {
        int row = m0 + wm + mt * 16 + j + half * 8;
        int col = n0 + wn + nt * 16 + r;
        float vv = (acc[mt][nt][j] + bias[col]) * mask[row];
        out[(size_t)row * N3 + col] = f2bf(vv);
      }
}

// ------------------------------------------------------- gated attention + residual
// kqv: [B*512][3072] bf16 with cols K|Q|V. One block = (b, h, 32-query slab).
__global__ __launch_bounds__(256) void attn(
    const unsigned short* __restrict__ kqv, const float* __restrict__ mask,
    const float* __restrict__ gate, const float* __restrict__ xin,
    unsigned short* __restrict__ xout)
{
  __shared__ float    sS[32 * 512];    // scores f32, 64KB
  __shared__ unsigned sQ[32 * 32];     // [q][dpair]   4KB
  __shared__ unsigned sKV[64 * 32];    // K/V chunk    8KB (reused)
  __shared__ unsigned sP[32 * 256];    // probs bf16  32KB, [q][keypair]
  __shared__ float    red[32 * 8];
  __shared__ float    gf[64];

  const int t = threadIdx.x, lane = t & 31, wv = t >> 5;
  const int half = lane >> 4, r = lane & 15;
  const int bid = blockIdx.x;
  const int qb = bid & 15;             // query slab (32 queries)
  const int h  = (bid >> 4) & 15;
  const int b  = bid >> 8;
  const size_t rowbase = (size_t)b * NSEQ;
  const int q0g = qb * 32;

  if (t < 64) gf[t] = 1.f + gate[b * D_ + h * 64 + t];
  __syncthreads();

  // Q slab, gated and pre-scaled by 1/sqrt(dh)=0.125
  #pragma unroll
  for (int s = 0; s < 4; s++) {
    int pp = t + s * 256;              // 1024 pairs = 32 q x 32 dpair
    int q = pp >> 5, dp = pp & 31;
    const unsigned short* src =
        kqv + (rowbase + q0g + q) * N3 + D_ + h * 64 + dp * 2;
    float f0 = bf2f(src[0]) * gf[dp * 2]     * 0.125f;
    float f1 = bf2f(src[1]) * gf[dp * 2 + 1] * 0.125f;
    sQ[q * 32 + dp] = pack2(f0, f1);
  }

  // ---- pass 1: scores, streaming 64-key K chunks
  for (int c = 0; c < 8; c++) {
    __syncthreads();
    #pragma unroll
    for (int s = 0; s < 8; s++) {      // 2048 pairs = 64 keys x 32 dpair
      int pp = t + s * 256;
      int kq = pp >> 5, dp = pp & 31;
      const unsigned short* src =
          kqv + (rowbase + c * 64 + kq) * N3 + h * 64 + dp * 2;
      float f0 = bf2f(src[0]) * gf[dp * 2];
      float f1 = bf2f(src[1]) * gf[dp * 2 + 1];
      sKV[kq * 32 + dp] = pack2(f0, f1);
    }
    __syncthreads();
    int qt = (wv >> 2) * 16, kt = (wv & 3) * 16;   // 2x4 tiles over 32q x 64k
    f32x8 acc; zero8(acc);
    #pragma unroll
    for (int kb = 0; kb < 64; kb += 32) {
      bf16x16 aF = ld_frag((const char*)sQ  + (qt + r) * 128 + kb * 2 + half * 16);
      bf16x16 bF = ld_frag((const char*)sKV + (kt + r) * 128 + kb * 2 + half * 16);
      acc = __builtin_amdgcn_wmma_f32_16x16x32_bf16(
          false, aF, false, bF, (short)0, acc, false, false);
    }
    #pragma unroll
    for (int j = 0; j < 8; j++) {
      int q = qt + j + half * 8;
      int key = c * 64 + kt + r;
      sS[q * 512 + key] = acc[j];
    }
  }
  __syncthreads();

  // ---- masked softmax over 512 keys, 8 threads per row
  {
    int row = t >> 3, seg = (t & 7) * 64;
    float mx = -3.0e38f;
    for (int k = 0; k < 64; k++) {
      int key = seg + k;
      if (mask[b * NSEQ + key] > 0.f) {
        float s = sS[row * 512 + key];
        mx = s > mx ? s : mx;
      }
    }
    red[row * 8 + (t & 7)] = mx;
    __syncthreads();
    float rmax = red[row * 8];
    #pragma unroll
    for (int i = 1; i < 8; i++) {
      float vv = red[row * 8 + i];
      rmax = vv > rmax ? vv : rmax;
    }
    __syncthreads();
    float sum = 0.f;
    for (int k = 0; k < 64; k++) {
      int key = seg + k;
      float e = (mask[b * NSEQ + key] > 0.f)
                    ? __expf(sS[row * 512 + key] - rmax) : 0.f;
      sS[row * 512 + key] = e;
      sum += e;
    }
    red[row * 8 + (t & 7)] = sum;
    __syncthreads();
    float rsum = 0.f;
    #pragma unroll
    for (int i = 0; i < 8; i++) rsum += red[row * 8 + i];
    float inv = 1.f / rsum;
    for (int k = 0; k < 32; k++) {     // 32 keypairs per segment
      int kp = (seg >> 1) + k;
      float e0 = sS[row * 512 + kp * 2]     * inv;
      float e1 = sS[row * 512 + kp * 2 + 1] * inv;
      sP[row * 256 + kp] = pack2(e0, e1);
    }
  }

  // ---- pass 2: O = P @ V, streaming transposed V chunks
  int qo = (wv >> 2) * 16, dt = (wv & 3) * 16;     // 2x4 tiles over 32q x 64d
  f32x8 oacc; zero8(oacc);
  for (int c = 0; c < 8; c++) {
    __syncthreads();
    #pragma unroll
    for (int s = 0; s < 8; s++) {      // 2048 dwords = 64 d x 32 keypair
      int pp = t + s * 256;
      int d = pp >> 5, kp = pp & 31;
      const unsigned short* src =
          kqv + (rowbase + c * 64 + kp * 2) * N3 + 2 * D_ + h * 64 + d;
      sKV[d * 32 + kp] = (unsigned)src[0] | ((unsigned)src[N3] << 16);
    }
    __syncthreads();
    #pragma unroll
    for (int kb = 0; kb < 64; kb += 32) {
      bf16x16 aF = ld_frag((const char*)sP  + (qo + r) * 1024 +
                           (c * 64 + kb) * 2 + half * 16);
      bf16x16 bF = ld_frag((const char*)sKV + (dt + r) * 128 + kb * 2 + half * 16);
      oacc = __builtin_amdgcn_wmma_f32_16x16x32_bf16(
          false, aF, false, bF, (short)0, oacc, false, false);
    }
  }
  #pragma unroll
  for (int j = 0; j < 8; j++) {        // residual add, bf16 store
    int q = q0g + qo + j + half * 8;
    int d = h * 64 + dt + r;
    size_t idx = (rowbase + q) * D_ + d;
    xout[idx] = f2bf(xin[idx] + oacc[j]);
  }
}

// ------------------------------------------------ output projection (bf16 A/B)
__global__ __launch_bounds__(256) void proj_gemm(
    const unsigned* __restrict__ Xd, const unsigned* __restrict__ Wt,
    const float* __restrict__ bias, float* __restrict__ out)
{
  __shared__ unsigned sA[2][128 * 16];
  __shared__ unsigned sB[2][128 * 16];
  const int t = threadIdx.x;
  const int lane = t & 31, wv = t >> 5;
  const int half = lane >> 4, r = lane & 15;
  const int m0 = blockIdx.x * 128;
  const int n0 = blockIdx.y * 128;
  const int wm = (wv >> 2) * 64;
  const int wn = (wv & 3) * 32;
  f32x8 acc[4][2];
  #pragma unroll
  for (int i = 0; i < 4; i++)
    #pragma unroll
    for (int j = 0; j < 2; j++) zero8(acc[i][j]);

  gemm_issue_tile(Xd, Wt, sA[0], sB[0], t, m0, n0, 0);
  wait_async();
  __syncthreads();
  int cur = 0;
  for (int kk = 0; kk < D_; kk += 32) {
    if (kk + 32 < D_)
      gemm_issue_tile(Xd, Wt, sA[cur ^ 1], sB[cur ^ 1], t, m0, n0, kk + 32);
    bf16x16 aF[4], bF[2];
    #pragma unroll
    for (int mt = 0; mt < 4; mt++)
      aF[mt] = ld_frag((const char*)sA[cur] + (wm + mt * 16 + r) * 64 + half * 16);
    #pragma unroll
    for (int nt = 0; nt < 2; nt++)
      bF[nt] = ld_frag((const char*)sB[cur] + (wn + nt * 16 + r) * 64 + half * 16);
    #pragma unroll
    for (int mt = 0; mt < 4; mt++)
      #pragma unroll
      for (int nt = 0; nt < 2; nt++)
        acc[mt][nt] = __builtin_amdgcn_wmma_f32_16x16x32_bf16(
            false, aF[mt], false, bF[nt], (short)0, acc[mt][nt], false, false);
    wait_async();
    __syncthreads();
    cur ^= 1;
  }
  #pragma unroll
  for (int mt = 0; mt < 4; mt++)
    #pragma unroll
    for (int nt = 0; nt < 2; nt++)
      #pragma unroll
      for (int j = 0; j < 8; j++) {
        int row = m0 + wm + mt * 16 + j + half * 8;
        int col = n0 + wn + nt * 16 + r;
        out[(size_t)row * D_ + col] = acc[mt][nt][j] + bias[col];
      }
}

// -------------------------------------------------------------------- launch
extern "C" void kernel_launch(void* const* d_in, const int* in_sizes, int n_in,
                              void* d_out, int out_size, void* d_ws, size_t ws_size,
                              hipStream_t stream) {
  const float* v      = (const float*)d_in[0];
  const float* tt     = (const float*)d_in[1];
  const float* v_mask = (const float*)d_in[2];
  const float* t_mask = (const float*)d_in[3];
  const float* W_v4t  = (const float*)d_in[4];
  const float* b_v4t  = (const float*)d_in[5];
  const float* W_t4v  = (const float*)d_in[6];
  const float* b_t4v  = (const float*)d_in[7];
  const float* W_v    = (const float*)d_in[8];
  const float* b_v    = (const float*)d_in[9];
  const float* W_t    = (const float*)d_in[10];
  const float* b_t    = (const float*)d_in[11];
  const float* W_vo   = (const float*)d_in[12];
  const float* b_vo   = (const float*)d_in[13];
  const float* W_to   = (const float*)d_in[14];
  const float* b_to   = (const float*)d_in[15];

  char* ws = (char*)d_ws;
  const size_t SMALL = 131072;                         // 32*1024*4
  const size_t KQVSZ = (size_t)MROWS * N3 * 2;         // 100663296
  const size_t XSZ   = (size_t)MROWS * D_ * 2;         // 33554432
  const size_t W3SZ  = (size_t)N3 * D_ * 2;            // 6291456
  const size_t W1SZ  = (size_t)D_ * D_ * 2;            // 2097152
  size_t off = 0;
  float* mean_v   = (float*)(ws + off); off += SMALL;
  float* mean_t   = (float*)(ws + off); off += SMALL;
  float* gate_v4t = (float*)(ws + off); off += SMALL;
  float* gate_t4v = (float*)(ws + off); off += SMALL;
  unsigned short* kqv_v = (unsigned short*)(ws + off); off += KQVSZ;
  unsigned short* kqv_t = (unsigned short*)(ws + off); off += KQVSZ;
  unsigned short* x_v   = (unsigned short*)(ws + off); off += XSZ;
  unsigned short* x_t   = (unsigned short*)(ws + off); off += XSZ;
  unsigned* xb_v  = (unsigned*)(ws + off); off += XSZ;   // relu(v) bf16
  unsigned* xb_t  = (unsigned*)(ws + off); off += XSZ;   // relu(t) bf16
  unsigned* wt_v  = (unsigned*)(ws + off); off += W3SZ;  // W_v  [n][kpair]
  unsigned* wt_t  = (unsigned*)(ws + off); off += W3SZ;  // W_t  [n][kpair]
  unsigned* wt_vo = (unsigned*)(ws + off); off += W1SZ;  // W_vo [n][kpair]
  unsigned* wt_to = (unsigned*)(ws + off); off += W1SZ;  // W_to [n][kpair]

  float* out_v = (float*)d_out;
  float* out_t = out_v + (size_t)MROWS * D_;

  // pre-passes: bf16 activations + transposed bf16 weights
  int cvtBlocks = (MROWS * D_) / (256 * 8);             // 8192
  cvt_relu_bf16<<<cvtBlocks, 256, 0, stream>>>(v,  xb_v);
  cvt_relu_bf16<<<cvtBlocks, 256, 0, stream>>>(tt, xb_t);
  transpose_w_bf16<<<dim3(N3 / 256, 512), 256, 0, stream>>>(W_v,  wt_v,  N3);
  transpose_w_bf16<<<dim3(N3 / 256, 512), 256, 0, stream>>>(W_t,  wt_t,  N3);
  transpose_w_bf16<<<dim3(D_ / 256, 512), 256, 0, stream>>>(W_vo, wt_vo, D_);
  transpose_w_bf16<<<dim3(D_ / 256, 512), 256, 0, stream>>>(W_to, wt_to, D_);

  mean_pool<<<256, 256, 0, stream>>>(v, tt, v_mask, t_mask, mean_v, mean_t);
  gate_gemm<<<256, 256, 0, stream>>>(mean_v, mean_t, W_v4t, b_v4t, W_t4v, b_t4v,
                                     gate_v4t, gate_t4v);
  dim3 g1(MROWS / 128, N3 / 128);
  kqv_gemm<<<g1, 256, 0, stream>>>(xb_v, wt_v, b_v, v_mask, kqv_v);
  kqv_gemm<<<g1, 256, 0, stream>>>(xb_t, wt_t, b_t, t_mask, kqv_t);
  // v_update uses t4v_gate; t_update uses v4t_gate
  attn<<<B_ * H_ * (NSEQ / 32), 256, 0, stream>>>(kqv_v, v_mask, gate_t4v, v,  x_v);
  attn<<<B_ * H_ * (NSEQ / 32), 256, 0, stream>>>(kqv_t, t_mask, gate_v4t, tt, x_t);
  dim3 g2(MROWS / 128, D_ / 128);
  proj_gemm<<<g2, 256, 0, stream>>>((const unsigned*)x_v, wt_vo, b_vo, out_v);
  proj_gemm<<<g2, 256, 0, stream>>>((const unsigned*)x_t, wt_to, b_to, out_t);
}
